// ALIGNNLike_2465311228414
// MI455X (gfx1250) — matrix operand order repaired and checked by reading, not compiled, not run on hardware
//
#include <hip/hip_runtime.h>
#include <hip/hip_bf16.h>

#define HDIM 128
#define NUM_GRAPHS 64
#define NLAYERS 3

typedef __attribute__((ext_vector_type(16))) __bf16 v16bf;
typedef __attribute__((ext_vector_type(8)))  float  v8f;

// ---------------------------------------------------------------------------
// degree / norm kernels
// ---------------------------------------------------------------------------
__global__ void degree_kernel(const int* __restrict__ src, const int* __restrict__ dst,
                              float* __restrict__ dsrc, float* __restrict__ ddst, int E) {
    int e = blockIdx.x * blockDim.x + threadIdx.x;
    if (e >= E) return;
    atomicAdd(&dsrc[src[e]], 1.0f);
    atomicAdd(&ddst[dst[e]], 1.0f);
}

__global__ void rsqrt_clamp_kernel(float* __restrict__ d, int n) {
    int i = blockIdx.x * blockDim.x + threadIdx.x;
    if (i >= n) return;
    float v = d[i];
    v = v < 1.0f ? 1.0f : v;
    d[i] = rsqrtf(v);
}

// ---------------------------------------------------------------------------
// feature init kernels
// ---------------------------------------------------------------------------
__global__ void embed_kernel(const int* __restrict__ z, const float* __restrict__ emb,
                             float* __restrict__ h, int N) {
    int i = blockIdx.x * blockDim.x + threadIdx.x;
    if (i >= N * HDIM) return;
    int node = i >> 7, f = i & (HDIM - 1);
    h[i] = emb[z[node] * HDIM + f];
}

__global__ void edge_init_kernel(const float* __restrict__ d, const float* __restrict__ w,
                                 const float* __restrict__ b, float* __restrict__ e, int N) {
    int i = blockIdx.x * blockDim.x + threadIdx.x;
    if (i >= N * HDIM) return;
    int node = i >> 7, f = i & (HDIM - 1);
    e[i] = d[node] * w[f] + b[f];
}

// ---------------------------------------------------------------------------
// scatter: agg[dst] += h[src] * out_norm[src]   (thread = one edge x float4)
// ---------------------------------------------------------------------------
__global__ void scatter_add_kernel(const float* __restrict__ hsrc, const int* __restrict__ src,
                                   const int* __restrict__ dst, const float* __restrict__ out_norm,
                                   float* __restrict__ agg, int E) {
    long long tid = (long long)blockIdx.x * blockDim.x + threadIdx.x;
    long long total = (long long)E * (HDIM / 4);
    if (tid >= total) return;
    int e = (int)(tid >> 5);          // HDIM/4 == 32
    int q = (int)(tid & 31);
    int s = src[e], t = dst[e];
    float on = out_norm[s];
    const float4 v = *(const float4*)(hsrc + (long long)s * HDIM + q * 4);
    float* p = agg + (long long)t * HDIM + q * 4;
    atomicAdd(p + 0, v.x * on);
    atomicAdd(p + 1, v.y * on);
    atomicAdd(p + 2, v.z * on);
    atomicAdd(p + 3, v.w * on);
}

// ---------------------------------------------------------------------------
// WMMA GEMM: Out = relu( (A * row_scale) @ W + bias ), A:[nrows,128], W:[128,128]
// 128 threads = 4 waves; block handles 64 rows (4 x 16-row tiles).
// W staged in LDS transposed as bf16 [n][k]; A tile staged bf16 [m][k] per wave.
// ---------------------------------------------------------------------------
__global__ __launch_bounds__(128)
void gemm_relu_wmma(const float* __restrict__ A, const float* __restrict__ row_scale,
                    const float* __restrict__ W, const float* __restrict__ bias,
                    float* __restrict__ Out, int nrows) {
    __shared__ __bf16 wlds[HDIM * HDIM];        // 32 KB, [n*128 + k] = W[k][n]
    __shared__ __bf16 alds[4][16 * HDIM];       // 16 KB, per-wave 16x128 tile

    const int tid  = threadIdx.x;
    const int wave = tid >> 5;
    const int lane = tid & 31;
    const int rowBlock = blockIdx.x * 64;
    const int waveRow0 = rowBlock + wave * 16;

    // stage W transposed, fp32 -> bf16
    for (int i = tid; i < HDIM * HDIM; i += 128) {
        int k = i >> 7, n = i & (HDIM - 1);
        wlds[n * HDIM + k] = (__bf16)W[i];
    }
    // stage this wave's A tile, scaled by in-norm, fp32 -> bf16
    for (int i = lane; i < 16 * HDIM; i += 32) {
        int m = i >> 7, k = i & (HDIM - 1);
        int r = waveRow0 + m;
        float v = 0.0f;
        if (r < nrows) v = A[(long long)r * HDIM + k] * row_scale[r];
        alds[wave][i] = (__bf16)v;
    }
    __syncthreads();

    const int half = lane >> 4;   // selects K-group per ISA layout
    const int ln   = lane & 15;

    v8f acc[8];
    v8f zero = {};
#pragma unroll
    for (int nt = 0; nt < 8; ++nt) acc[nt] = zero;

#pragma unroll
    for (int kt = 0; kt < 4; ++kt) {
        // A frag (16x32 bf16): m = ln; elems 0-7: K = kt*32 + 8*half + j
        //                              elems 8-15: K = kt*32 + 16 + 8*half + j
        union { v16bf v; uint4 u[2]; } af;
        const __bf16* abase = &alds[wave][ln * HDIM + kt * 32];
        af.u[0] = *(const uint4*)(abase + 8 * half);
        af.u[1] = *(const uint4*)(abase + 16 + 8 * half);
#pragma unroll
        for (int nt = 0; nt < 8; ++nt) {
            // B frag (32x16 bf16): n = nt*16 + ln; elems 0-7: K = kt*32 + 8*half + j, elems 8-15: K = kt*32 + 16 + 8*half + j
            union { v16bf v; uint4 u[2]; } bfg;
            const __bf16* bbase = &wlds[(nt * 16 + ln) * HDIM + kt * 32 + 8 * half];
            bfg.u[0] = *(const uint4*)(bbase);
            bfg.u[1] = *(const uint4*)(bbase + 16);
            acc[nt] = __builtin_amdgcn_wmma_f32_16x16x32_bf16(
                false, af.v, false, bfg.v, (short)0, acc[nt], false, false);
        }
    }

    // epilogue: C layout -> element r of v8f: m = r + 8*half, n = ln (+16*nt)
#pragma unroll
    for (int nt = 0; nt < 8; ++nt) {
#pragma unroll
        for (int r = 0; r < 8; ++r) {
            int m   = r + 8 * half;
            int row = waveRow0 + m;
            int n   = nt * 16 + ln;
            if (row < nrows) {
                float v = acc[nt][r] + bias[n];
                Out[(long long)row * HDIM + n] = v > 0.0f ? v : 0.0f;
            }
        }
    }
}

// ---------------------------------------------------------------------------
// mean pooling (accumulate phase)
// ---------------------------------------------------------------------------
__global__ void pool_kernel(const float* __restrict__ h, const int* __restrict__ seg,
                            float* __restrict__ sums, float* __restrict__ cnts, int N) {
    int i = blockIdx.x * blockDim.x + threadIdx.x;
    if (i >= N * HDIM) return;
    int node = i >> 7, f = i & (HDIM - 1);
    int g = seg[node];
    atomicAdd(&sums[g * HDIM + f], h[i]);
    if (f == 0) atomicAdd(&cnts[g], 1.0f);
}

// ---------------------------------------------------------------------------
// head: x = [hg, he]; y = silu(x @ h1_w + h1_b); out = y @ h2_w + h2_b
// grid = NUM_GRAPHS blocks x 128 threads
// ---------------------------------------------------------------------------
__global__ __launch_bounds__(128)
void head_kernel(const float* __restrict__ gsum, const float* __restrict__ gcnt,
                 const float* __restrict__ esum, const float* __restrict__ ecnt,
                 const float* __restrict__ h1w, const float* __restrict__ h1b,
                 const float* __restrict__ h2w, const float* __restrict__ h2b,
                 float* __restrict__ out) {
    __shared__ float red[128];
    int g = blockIdx.x;
    int j = threadIdx.x;
    float gc = gcnt[g] < 1.0f ? 1.0f : gcnt[g];
    float ec = ecnt[g] < 1.0f ? 1.0f : ecnt[g];
    float acc = h1b[j];
    for (int k = 0; k < HDIM; ++k)
        acc += (gsum[g * HDIM + k] / gc) * h1w[k * HDIM + j];
    for (int k = 0; k < HDIM; ++k)
        acc += (esum[g * HDIM + k] / ec) * h1w[(HDIM + k) * HDIM + j];
    float y = acc / (1.0f + __expf(-acc));   // silu
    red[j] = y * h2w[j];
    __syncthreads();
    for (int s = 64; s > 0; s >>= 1) {
        if (j < s) red[j] += red[j + s];
        __syncthreads();
    }
    if (j == 0) out[g] = red[0] + h2b[0];
}

// ---------------------------------------------------------------------------
// launcher
// ---------------------------------------------------------------------------
static inline char* carve(char*& p, size_t bytes) {
    char* r = p;
    p += (bytes + 255) & ~(size_t)255;
    return r;
}

extern "C" void kernel_launch(void* const* d_in, const int* in_sizes, int n_in,
                              void* d_out, int out_size, void* d_ws, size_t ws_size,
                              hipStream_t stream) {
    const int*   z      = (const int*)  d_in[0];
    const float* dfeat  = (const float*)d_in[1];
    const int*   g_src  = (const int*)  d_in[2];
    const int*   g_dst  = (const int*)  d_in[3];
    const int*   lg_src = (const int*)  d_in[4];
    const int*   lg_dst = (const int*)  d_in[5];
    const int*   n2g    = (const int*)  d_in[6];
    const int*   ln2g   = (const int*)  d_in[7];
    const float* emb    = (const float*)d_in[8];
    const float* eprojw = (const float*)d_in[9];
    const float* eprojb = (const float*)d_in[10];
    const float* gW     = (const float*)d_in[11];
    const float* gB     = (const float*)d_in[12];
    const float* lW     = (const float*)d_in[13];
    const float* lB     = (const float*)d_in[14];
    const float* h1w    = (const float*)d_in[15];
    const float* h1b    = (const float*)d_in[16];
    const float* h2w    = (const float*)d_in[17];
    const float* h2b    = (const float*)d_in[18];

    const int N_G  = in_sizes[0];
    const int E_G  = in_sizes[2];
    const int E_LG = in_sizes[4];
    const int N_LG = in_sizes[7];

    char* ws = (char*)d_ws;
    float* eA    = (float*)carve(ws, (size_t)N_LG * HDIM * 4);
    float* eB    = (float*)carve(ws, (size_t)N_LG * HDIM * 4);
    float* hA    = (float*)carve(ws, (size_t)N_G  * HDIM * 4);
    float* hB    = (float*)carve(ws, (size_t)N_G  * HDIM * 4);
    float* g_on  = (float*)carve(ws, (size_t)N_G  * 4);
    float* g_in  = (float*)carve(ws, (size_t)N_G  * 4);
    float* lg_on = (float*)carve(ws, (size_t)N_LG * 4);
    float* lg_in = (float*)carve(ws, (size_t)N_LG * 4);
    float* gsum  = (float*)carve(ws, (size_t)NUM_GRAPHS * HDIM * 4);
    float* esum  = (float*)carve(ws, (size_t)NUM_GRAPHS * HDIM * 4);
    float* gcnt  = (float*)carve(ws, (size_t)NUM_GRAPHS * 4);
    float* ecnt  = (float*)carve(ws, (size_t)NUM_GRAPHS * 4);

    const int TB = 256;
    auto blks = [](long long n, int tb) { return (unsigned)((n + tb - 1) / tb); };

    // ---- degrees & norms (computed once, reused across layers) ----
    hipMemsetAsync(g_on,  0, (size_t)N_G  * 4, stream);
    hipMemsetAsync(g_in,  0, (size_t)N_G  * 4, stream);
    hipMemsetAsync(lg_on, 0, (size_t)N_LG * 4, stream);
    hipMemsetAsync(lg_in, 0, (size_t)N_LG * 4, stream);
    degree_kernel<<<blks(E_G, TB),  TB, 0, stream>>>(g_src,  g_dst,  g_on,  g_in,  E_G);
    degree_kernel<<<blks(E_LG, TB), TB, 0, stream>>>(lg_src, lg_dst, lg_on, lg_in, E_LG);
    rsqrt_clamp_kernel<<<blks(N_G, TB),  TB, 0, stream>>>(g_on,  N_G);
    rsqrt_clamp_kernel<<<blks(N_G, TB),  TB, 0, stream>>>(g_in,  N_G);
    rsqrt_clamp_kernel<<<blks(N_LG, TB), TB, 0, stream>>>(lg_on, N_LG);
    rsqrt_clamp_kernel<<<blks(N_LG, TB), TB, 0, stream>>>(lg_in, N_LG);

    // ---- feature init ----
    embed_kernel<<<blks((long long)N_G * HDIM, TB), TB, 0, stream>>>(z, emb, hA, N_G);
    edge_init_kernel<<<blks((long long)N_LG * HDIM, TB), TB, 0, stream>>>(dfeat, eprojw, eprojb, eA, N_LG);

    // ---- atom-graph GraphConv layers ----
    for (int l = 0; l < NLAYERS; ++l) {
        hipMemsetAsync(hB, 0, (size_t)N_G * HDIM * 4, stream);
        scatter_add_kernel<<<blks((long long)E_G * (HDIM / 4), TB), TB, 0, stream>>>(
            hA, g_src, g_dst, g_on, hB, E_G);
        gemm_relu_wmma<<<(N_G + 63) / 64, 128, 0, stream>>>(
            hB, g_in, gW + (size_t)l * HDIM * HDIM, gB + (size_t)l * HDIM, hA, N_G);
    }

    // ---- line-graph GraphConv layers ----
    for (int l = 0; l < NLAYERS; ++l) {
        hipMemsetAsync(eB, 0, (size_t)N_LG * HDIM * 4, stream);
        scatter_add_kernel<<<blks((long long)E_LG * (HDIM / 4), TB), TB, 0, stream>>>(
            eA, lg_src, lg_dst, lg_on, eB, E_LG);
        gemm_relu_wmma<<<(N_LG + 63) / 64, 128, 0, stream>>>(
            eB, lg_in, lW + (size_t)l * HDIM * HDIM, lB + (size_t)l * HDIM, eA, N_LG);
    }

    // ---- pooling + head ----
    hipMemsetAsync(gsum, 0, (size_t)NUM_GRAPHS * HDIM * 4, stream);
    hipMemsetAsync(esum, 0, (size_t)NUM_GRAPHS * HDIM * 4, stream);
    hipMemsetAsync(gcnt, 0, (size_t)NUM_GRAPHS * 4, stream);
    hipMemsetAsync(ecnt, 0, (size_t)NUM_GRAPHS * 4, stream);
    pool_kernel<<<blks((long long)N_G * HDIM, TB),  TB, 0, stream>>>(hA, n2g,  gsum, gcnt, N_G);
    pool_kernel<<<blks((long long)N_LG * HDIM, TB), TB, 0, stream>>>(eA, ln2g, esum, ecnt, N_LG);
    head_kernel<<<NUM_GRAPHS, 128, 0, stream>>>(gsum, gcnt, esum, ecnt,
                                                h1w, h1b, h2w, h2b, (float*)d_out);
}
